// decoderHead_15152644621126
// MI455X (gfx1250) — matrix operand order repaired and checked by reading, not compiled, not run on hardware
//
#include <hip/hip_runtime.h>
#include <hip/hip_bf16.h>

#define B_  4
#define S_  4096
#define E_  512
#define DK_ 64

#define KSTRIDE 88          // bf16 elems per LDS K/V row: 128B data + 48B TDM pad
#define PSTR    40          // bf16 elems per pbuf row (20 dwords, conflict-free)
#define NBLK    (S_ / 64)   // 64-key blocks

typedef __attribute__((ext_vector_type(16))) __bf16 v16bf;
typedef __attribute__((ext_vector_type(8)))  float  v8f;
typedef unsigned int uint32x4 __attribute__((ext_vector_type(4)));
typedef int          int32x4  __attribute__((ext_vector_type(4)));
typedef int          int32x8  __attribute__((ext_vector_type(8)));

#if defined(__gfx1250__) && __has_builtin(__builtin_amdgcn_tensor_load_to_lds)
#define HAVE_TDM 1
#else
#define HAVE_TDM 0
#endif

static __device__ __forceinline__ v8f wmma_bf16(v16bf a, v16bf b, v8f c) {
  return __builtin_amdgcn_wmma_f32_16x16x32_bf16(false, a, false, b, (short)0, c,
                                                 false, false);
}

// TDM 2D tile load: tile = 32 dwords x 64 rows, row padded 128B->176B in LDS.
static __device__ __forceinline__ void tdm_load_2d(unsigned lds_addr,
                                                   unsigned long long gaddr,
                                                   unsigned tensor_d0,
                                                   unsigned stride0_dw) {
#if HAVE_TDM
  uint32x4 g0;
  g0[0] = 1u;                                       // count=1 valid descriptor
  g0[1] = lds_addr;                                 // LDS byte address
  g0[2] = (unsigned)(gaddr & 0xffffffffu);          // global_addr[31:0]
  g0[3] = (unsigned)((gaddr >> 32) & 0x01ffffffu)   // global_addr[56:32]
        | 0x80000000u;                              // type=2 (image)
  int32x8 g1;
  // data_size=4B (2), pad_enable, pad_interval=32dw (4), pad_amount=12dw (11)
  g1[0] = (int)((2u << 16) | (1u << 20) | (4u << 22) | (11u << 25));
  g1[1] = (int)((tensor_d0 & 0xffffu) << 16);                       // dim0 lo16
  g1[2] = (int)(((tensor_d0 >> 16) & 0xffffu) | (64u << 16));       // dim0 hi / dim1=64
  g1[3] = (int)(32u << 16);                                         // tile_dim0=32 dw
  g1[4] = (int)64u;                                                 // tile_dim1=64 rows
  g1[5] = (int)stride0_dw;                                          // dim0 stride lo
  g1[6] = 0;
  g1[7] = 0;
  int32x4 z4 = {};
#if __clang_major__ >= 23
  int32x8 z8 = {};
  __builtin_amdgcn_tensor_load_to_lds(g0, g1, z4, z4, z8, 0);
#else
  __builtin_amdgcn_tensor_load_to_lds(g0, g1, z4, z4, 0);
#endif
#else
  (void)lds_addr; (void)gaddr; (void)tensor_d0; (void)stride0_dw;
#endif
}

// ---------------------------------------------------------------------------
// Kernel 1: transpose weights -> bf16 Wt[3][64][512], zero lengths
// ---------------------------------------------------------------------------
__global__ void prep_kernel(const float* __restrict__ Wq,
                            const float* __restrict__ Wk,
                            const float* __restrict__ Wv,
                            __bf16* __restrict__ Wt, int* __restrict__ lengths) {
  int t = blockIdx.x * blockDim.x + threadIdx.x;
  if (t < B_) lengths[t] = 0;
  if (t < 3 * DK_ * E_) {
    int m = t / (DK_ * E_);
    int r = t % (DK_ * E_);
    int n = r / E_;
    int k = r % E_;
    const float* W = (m == 0) ? Wq : ((m == 1) ? Wk : Wv);
    Wt[t] = (__bf16)W[k * DK_ + n];
  }
}

// ---------------------------------------------------------------------------
// Kernel 2: pad mask. One wave per row.
// ---------------------------------------------------------------------------
__global__ void mask_kernel(const float* __restrict__ x, float* __restrict__ mask) {
  int wave = (blockIdx.x * blockDim.x + threadIdx.x) >> 5;
  int lane = threadIdx.x & 31;
  if (wave >= B_ * S_) return;
  const float* row = x + (size_t)wave * E_;
  int nz = 0;
  for (int i = lane; i < E_; i += 32) nz |= (row[i] != 0.0f) ? 1 : 0;
  nz = __any(nz);
  if (lane == 0) mask[wave] = nz ? 1.0f : 0.0f;
}

// ---------------------------------------------------------------------------
// Kernel 3: QKV projection (unchanged: clean b128 + wmma codegen).
// ---------------------------------------------------------------------------
__global__ void __launch_bounds__(128) qkv_kernel(
    const float* __restrict__ x, const float* __restrict__ ctx,
    const __bf16* __restrict__ Wt, const float* __restrict__ mask,
    __bf16* __restrict__ Qb, __bf16* __restrict__ Kb, __bf16* __restrict__ Vt) {
  const int lane = threadIdx.x & 31;
  const int wave = threadIdx.x >> 5;
  const int half = lane >> 4;
  const int lm   = lane & 15;
  const int rowBase = (blockIdx.x * 4 + wave) * 16;

  v8f zero = {};
  v8f acc[12];
  #pragma unroll
  for (int i = 0; i < 12; ++i) acc[i] = zero;

  const float* xrow = x   + (size_t)(rowBase + lm) * E_;
  const float* crow = ctx + (size_t)(rowBase + lm) * E_;

  for (int ks = 0; ks < E_ / 32; ++ks) {
    const int kb = ks * 32;
    v16bf ax, ac;
    #pragma unroll
    for (int v = 0; v < 8; ++v) {
      int kk = kb + ((v < 4) ? 0 : 16) + half * 8 + (v & 3) * 2;
      ax[2*v]   = (__bf16)xrow[kk];
      ax[2*v+1] = (__bf16)xrow[kk+1];
      ac[2*v]   = (__bf16)crow[kk];
      ac[2*v+1] = (__bf16)crow[kk+1];
    }
    #pragma unroll
    for (int mtx = 0; mtx < 3; ++mtx) {
      #pragma unroll
      for (int nt = 0; nt < 4; ++nt) {
        const __bf16* wcol = Wt + (size_t)mtx * DK_ * E_ + (size_t)(nt*16 + lm) * E_;
        v16bf bw;
        #pragma unroll
        for (int v = 0; v < 8; ++v) {
          int kk = kb + ((v < 4) ? 0 : 16) + half * 8 + (v & 3) * 2;
          bw[2*v]   = wcol[kk];
          bw[2*v+1] = wcol[kk+1];
        }
        acc[mtx*4 + nt] = wmma_bf16((mtx == 0) ? ax : ac, bw, acc[mtx*4 + nt]);
      }
    }
  }

  float mr[8];
  #pragma unroll
  for (int r = 0; r < 8; ++r) mr[r] = mask[rowBase + r + half * 8];

  const int batch   = rowBase / S_;
  const int keyBase = rowBase % S_;
  #pragma unroll
  for (int nt = 0; nt < 4; ++nt) {
    int d = nt * 16 + lm;
    #pragma unroll
    for (int r = 0; r < 8; ++r) {
      int row = rowBase + r + half * 8;
      Qb[(size_t)row * DK_ + d] = (__bf16)(acc[0*4 + nt][r] * mr[r]);
      Kb[(size_t)row * DK_ + d] = (__bf16)(acc[1*4 + nt][r] * mr[r]);
      Vt[(size_t)batch * DK_ * S_ + (size_t)d * S_ + keyBase + r + half * 8] =
          (__bf16)(acc[2*4 + nt][r] * mr[r]);
    }
  }
}

// ---------------------------------------------------------------------------
// Kernel 4: lengths[b] = count of nonzero in energy[b,0,:]
// ---------------------------------------------------------------------------
__global__ void lengths_kernel(const __bf16* __restrict__ Qb,
                               const __bf16* __restrict__ Kb,
                               int* __restrict__ lengths) {
  int b = blockIdx.x;
  float q0[DK_];
  const __bf16* qrow = Qb + (size_t)b * S_ * DK_;
  #pragma unroll
  for (int d = 0; d < DK_; ++d) q0[d] = (float)qrow[d];
  int cnt = 0;
  for (int k = threadIdx.x; k < S_; k += blockDim.x) {
    const __bf16* krow = Kb + ((size_t)b * S_ + k) * DK_;
    float dot = 0.f;
    #pragma unroll
    for (int d = 0; d < DK_; ++d) dot += q0[d] * (float)krow[d];
    cnt += (dot != 0.0f) ? 1 : 0;
  }
  atomicAdd(&lengths[b], cnt);
}

// ---------------------------------------------------------------------------
// Kernel 5: flash attention with TDM double-buffered K/V staging in LDS.
// Block = 8 waves = 8 query tiles of one batch; wave 0 drives the TDM.
// ---------------------------------------------------------------------------
__global__ void __launch_bounds__(256) attn_kernel(
    const __bf16* __restrict__ Qb, const __bf16* __restrict__ Kb,
    const __bf16* __restrict__ Vt, const float* __restrict__ mask,
    const int* __restrict__ lengths, float* __restrict__ out) {
  __shared__ __bf16 kbuf[2][64 * KSTRIDE];   // [key][d], padded rows
  __shared__ __bf16 vbuf[2][64 * KSTRIDE];   // [d][key], padded rows
  __shared__ __bf16 pbuf[8][16 * PSTR];      // per-wave P transpose staging

  const int lane  = threadIdx.x & 31;
  const int wave  = threadIdx.x >> 5;
  const int half  = lane >> 4;
  const int lm    = lane & 15;
  const int tile  = blockIdx.x * 8 + wave;
  const int batch = tile >> 8;            // 256 query tiles per batch
  const int qBase = (tile & 255) * 16;
  const int len   = lengths[batch];

  const __bf16* kg = Kb + (size_t)batch * S_ * DK_;   // [key][d]
  const __bf16* vg = Vt + (size_t)batch * DK_ * S_;   // [d][key]

  // Q A-fragments, resident for the whole key loop
  v16bf aq[2];
  const __bf16* qrow = Qb + ((size_t)batch * S_ + qBase + lm) * DK_;
  #pragma unroll
  for (int f = 0; f < 2; ++f)
    #pragma unroll
    for (int v = 0; v < 8; ++v) {
      int dd = f * 32 + ((v < 4) ? 0 : 16) + half * 8 + (v & 3) * 2;
      aq[f][2*v]   = qrow[dd];
      aq[f][2*v+1] = qrow[dd+1];
    }

  v8f zero = {};
  v8f oacc[4];
  #pragma unroll
  for (int i = 0; i < 4; ++i) oacc[i] = zero;
  float rmax[8], rsum[8];
  #pragma unroll
  for (int r = 0; r < 8; ++r) { rmax[r] = -__builtin_inff(); rsum[r] = 0.0f; }

#if HAVE_TDM
  // prologue: wave 0 stages block 0 into buffer 0 via the Tensor Data Mover
  if (wave == 0) {
    tdm_load_2d((unsigned)(size_t)&kbuf[0][0],
                (unsigned long long)(size_t)kg, /*dim0=*/32, /*stride0=*/32);
    tdm_load_2d((unsigned)(size_t)&vbuf[0][0],
                (unsigned long long)(size_t)vg, /*dim0=*/S_/2, /*stride0=*/S_/2);
  }
#endif

  for (int blk = 0; blk < NBLK; ++blk) {
    const int cur = blk & 1;
#if HAVE_TDM
    if (wave == 0) __builtin_amdgcn_s_wait_tensorcnt(0);
    __syncthreads();                      // buf[cur] ready; buf[cur^1] free
    if (wave == 0 && blk + 1 < NBLK) {
      const int nb = (blk + 1) * 64;
      tdm_load_2d((unsigned)(size_t)&kbuf[cur ^ 1][0],
                  (unsigned long long)(size_t)(kg + (size_t)nb * DK_), 32, 32);
      tdm_load_2d((unsigned)(size_t)&vbuf[cur ^ 1][0],
                  (unsigned long long)(size_t)(vg + nb), S_/2, S_/2);
    }
#else
    __syncthreads();
    // cooperative fallback copy: 64 rows x 128B each, b128 chunks
    for (int c = threadIdx.x; c < 512; c += 256) {
      int row = c >> 3, col = c & 7;
      *(int4*)&kbuf[cur][row * KSTRIDE + col * 8] =
          *(const int4*)(kg + ((size_t)blk * 64 + row) * DK_ + col * 8);
      *(int4*)&vbuf[cur][row * KSTRIDE + col * 8] =
          *(const int4*)(vg + (size_t)row * S_ + blk * 64 + col * 8);
    }
    __syncthreads();
#endif

    const __bf16* kl = kbuf[cur];
    const __bf16* vl = vbuf[cur];

    #pragma unroll
    for (int sub = 0; sub < 2; ++sub) {
      const int ks = sub * 32;            // local key offset within block
      const int kb = blk * 64 + ks;       // global key base

      // energy: 16 queries x 32 keys, B-fragments from LDS
      v8f e0 = zero, e1 = zero;
      #pragma unroll
      for (int f = 0; f < 2; ++f) {
        const __bf16* krow0 = kl + (size_t)(ks + lm) * KSTRIDE;
        const __bf16* krow1 = kl + (size_t)(ks + 16 + lm) * KSTRIDE;
        v16bf bk0, bk1;
        #pragma unroll
        for (int v = 0; v < 8; ++v) {
          int dd = f * 32 + ((v < 4) ? 0 : 16) + half * 8 + (v & 3) * 2;
          bk0[2*v] = krow0[dd]; bk0[2*v+1] = krow0[dd+1];
          bk1[2*v] = krow1[dd]; bk1[2*v+1] = krow1[dd+1];
        }
        e0 = wmma_bf16(aq[f], bk0, e0);
        e1 = wmma_bf16(aq[f], bk1, e1);
      }

      // online softmax (row state replicated across each 16-lane half)
      bool c0 = (kb + lm) < len, c1 = (kb + 16 + lm) < len;
      float p0[8], p1[8], sc[8];
      #pragma unroll
      for (int r = 0; r < 8; ++r) {
        float a0 = c0 ? e0[r] * 0.125f : -__builtin_inff();
        float a1 = c1 ? e1[r] * 0.125f : -__builtin_inff();
        float m = fmaxf(a0, a1);
        #pragma unroll
        for (int off = 1; off < 16; off <<= 1) m = fmaxf(m, __shfl_xor(m, off, 16));
        float mnew = fmaxf(rmax[r], m);
        float s  = (rmax[r] > -__builtin_inff()) ? __expf(rmax[r] - mnew) : 0.0f;
        float q0 = (c0 && mnew > -__builtin_inff()) ? __expf(a0 - mnew) : 0.0f;
        float q1 = (c1 && mnew > -__builtin_inff()) ? __expf(a1 - mnew) : 0.0f;
        float ps = q0 + q1;
        #pragma unroll
        for (int off = 1; off < 16; off <<= 1) ps += __shfl_xor(ps, off, 16);
        rsum[r] = rsum[r] * s + ps;
        rmax[r] = mnew;
        sc[r] = s; p0[r] = q0; p1[r] = q1;
      }
      #pragma unroll
      for (int nt = 0; nt < 4; ++nt)
        #pragma unroll
        for (int r = 0; r < 8; ++r) oacc[nt][r] *= sc[r];

      // stage P (D layout) -> LDS -> reload as A fragment (wave-private rows)
      #pragma unroll
      for (int r = 0; r < 8; ++r) {
        int m = r + half * 8;
        pbuf[wave][m * PSTR + lm]      = (__bf16)p0[r];
        pbuf[wave][m * PSTR + 16 + lm] = (__bf16)p1[r];
      }
      v16bf ap;
      #pragma unroll
      for (int v = 0; v < 8; ++v) {
        int kk = ((v < 4) ? 0 : 16) + half * 8 + (v & 3) * 2;
        ap[2*v]   = pbuf[wave][lm * PSTR + kk];
        ap[2*v+1] = pbuf[wave][lm * PSTR + kk + 1];
      }

      // P @ V: contraction over 32 keys, V B-fragments from LDS
      #pragma unroll
      for (int nt = 0; nt < 4; ++nt) {
        const __bf16* vrow = vl + (size_t)(nt * 16 + lm) * KSTRIDE + ks;
        v16bf bv;
        #pragma unroll
        for (int v = 0; v < 8; ++v) {
          int kk = ((v < 4) ? 0 : 16) + half * 8 + (v & 3) * 2;
          bv[2*v] = vrow[kk]; bv[2*v+1] = vrow[kk+1];
        }
        oacc[nt] = wmma_bf16(ap, bv, oacc[nt]);
      }
    }
  }

  // epilogue: /rowsum, *query mask, store f32
  float inv[8];
  #pragma unroll
  for (int r = 0; r < 8; ++r) {
    float qm = mask[(size_t)batch * S_ + qBase + r + half * 8];
    inv[r] = (rsum[r] > 0.0f) ? qm / rsum[r] : 0.0f;
  }
  #pragma unroll
  for (int nt = 0; nt < 4; ++nt) {
    int d = nt * 16 + lm;
    #pragma unroll
    for (int r = 0; r < 8; ++r) {
      int q = qBase + r + half * 8;
      out[((size_t)batch * S_ + q) * DK_ + d] = oacc[nt][r] * inv[r];
    }
  }
}

// ---------------------------------------------------------------------------
extern "C" void kernel_launch(void* const* d_in, const int* in_sizes, int n_in,
                              void* d_out, int out_size, void* d_ws, size_t ws_size,
                              hipStream_t stream) {
  const float* x  = (const float*)d_in[0];
  const float* cx = (const float*)d_in[1];
  const float* Wq = (const float*)d_in[2];
  const float* Wk = (const float*)d_in[3];
  const float* Wv = (const float*)d_in[4];
  float* out = (float*)d_out;

  char* ws = (char*)d_ws;
  size_t off = 0;
  auto alloc = [&](size_t bytes) -> void* {
    void* p = ws + off;
    off = (off + bytes + 255) & ~(size_t)255;
    return p;
  };
  __bf16* Wt   = (__bf16*)alloc((size_t)3 * DK_ * E_ * 2);
  __bf16* Qb   = (__bf16*)alloc((size_t)B_ * S_ * DK_ * 2);
  __bf16* Kb   = (__bf16*)alloc((size_t)B_ * S_ * DK_ * 2);
  __bf16* Vt   = (__bf16*)alloc((size_t)B_ * S_ * DK_ * 2);
  float*  mask = (float*)alloc((size_t)B_ * S_ * 4);
  int* lengths = (int*)alloc(B_ * sizeof(int));

  prep_kernel<<<(3 * DK_ * E_ + 255) / 256, 256, 0, stream>>>(Wq, Wk, Wv, Wt, lengths);
  mask_kernel<<<(B_ * S_) / 8, 256, 0, stream>>>(x, mask);
  qkv_kernel<<<(B_ * S_) / 64, 128, 0, stream>>>(x, cx, Wt, mask, Qb, Kb, Vt);
  lengths_kernel<<<B_, 256, 0, stream>>>(Qb, Kb, lengths);
  attn_kernel<<<(B_ * S_ / 16) / 8, 256, 0, stream>>>(Qb, Kb, Vt, mask, lengths, out);
}
